// MegNet_83708912599383
// MI455X (gfx1250) — matrix operand order repaired and checked by reading, not compiled, not run on hardware
//
#include <hip/hip_runtime.h>
#include <cstdint>
#include <cstddef>

// ---------------------------------------------------------------------------
// Types / helpers
// ---------------------------------------------------------------------------
typedef __attribute__((ext_vector_type(8)))  float        v8f;
typedef __attribute__((ext_vector_type(16))) __bf16       v16bf;
typedef __attribute__((ext_vector_type(4)))  unsigned int uint4v;
typedef __attribute__((ext_vector_type(2)))  unsigned int uint2v;
typedef __attribute__((ext_vector_type(4)))  float        float4v;

// Wave-local LDS fence: DS ops from one wave complete in order; wait + compiler
// barrier makes cross-lane store->load handoffs inside a wave safe.
#define LDS_FENCE() asm volatile("s_wait_dscnt 0x0" ::: "memory")

__device__ __forceinline__ unsigned short f2bf(float f) {
    unsigned int u = __float_as_uint(f);
    u += 0x7FFFu + ((u >> 16) & 1u);   // round-to-nearest-even
    return (unsigned short)(u >> 16);
}
// Branch-free SELU: exp is computed unconditionally (clamped to <=0 so the
// discarded lane value can't overflow); selection lowers to v_cndmask instead
// of s_and_saveexec EXEC-mask branch chains.
__device__ __forceinline__ float selu_f(float x) {
    const float sc = 1.0507009873554805f, al = 1.6732632423543772f;
    float e   = __expf(fminf(x, 0.f));
    float neg = sc * al * (e - 1.f);
    float pos = sc * x;
    return x > 0.f ? pos : neg;
}
__device__ __forceinline__ float sigm_f(float x) { return 1.f / (1.f + __expf(-x)); }

// ---------------------------------------------------------------------------
// Wave-level bf16 WMMA GEMM pieces (wave32, 16-row tile per wave)
// ---------------------------------------------------------------------------
// xt : LDS bf16 activation tile [16][xs]   (row = M)
// wt : LDS bf16 weight tile, transposed: [N][ws] with wt[n][k] = W[k][n]
// A fragment (16-bit A 16x32): lanes 0-15 -> M=lane, K in {0..7,16..23};
//                              lanes 16-31 -> M=lane-16, K in {8..15,24..31}
// B fragment: lanes 0-15 -> N=lane, K in {0..7,16..23}; lanes 16-31 -> K in {8..15,24..31}
// D (f32 16x16): vgpr i, lanes 0-15 -> (M=i, N=lane); lanes 16-31 -> (M=8+i)
template<int K, int N>
__device__ __forceinline__ void wave_gemm(const unsigned short* xt, int xs,
                                          const unsigned short* wt, int wss,
                                          const float* bias, float out[][8]) {
    const int lane = threadIdx.x & 31;
    const int half = lane >> 4;
    const int mn   = lane & 15;
    const unsigned short* xrow = xt + mn * xs;
#pragma unroll
    for (int nt = 0; nt < N / 16; ++nt) {
        const unsigned short* wrow = wt + (nt * 16 + mn) * wss;
        v8f acc;
#pragma unroll
        for (int i = 0; i < 8; ++i) acc[i] = 0.f;
#pragma unroll
        for (int kt = 0; kt < K / 32; ++kt) {
            union { v16bf v; uint4v u[2]; } af, bfg;
            const int ca = kt * 32 + half * 8;
            af.u[0] = *(const uint4v*)(xrow + ca);
            af.u[1] = *(const uint4v*)(xrow + ca + 16);
            const int cb = kt * 32 + half * 8;
            bfg.u[0] = *(const uint4v*)(wrow + cb);
            bfg.u[1] = *(const uint4v*)(wrow + cb + 16);
            acc = __builtin_amdgcn_wmma_f32_16x16x32_bf16(false, af.v, false, bfg.v,
                                                          (short)0, acc, false, false);
        }
        const float b = bias[nt * 16 + mn];
#pragma unroll
        for (int i = 0; i < 8; ++i) out[nt][i] = selu_f(acc[i] + b);
    }
}

template<int N>
__device__ __forceinline__ void wave_store_tile(unsigned short* yt, int ys,
                                                const float out[][8]) {
    const int lane = threadIdx.x & 31;
    const int half = lane >> 4, n0 = lane & 15;
#pragma unroll
    for (int nt = 0; nt < N / 16; ++nt)
#pragma unroll
        for (int i = 0; i < 8; ++i)
            yt[(half * 8 + i) * ys + nt * 16 + n0] = f2bf(out[nt][i]);
}

template<int N>
__device__ __forceinline__ void wave_store_global(float* g, int row0, int rows,
                                                  const float out[][8]) {
    const int lane = threadIdx.x & 31;
    const int half = lane >> 4, n0 = lane & 15;
#pragma unroll
    for (int i = 0; i < 8; ++i) {
        int r = row0 + half * 8 + i;
        if (r < rows) {
#pragma unroll
            for (int nt = 0; nt < N / 16; ++nt)
                g[(size_t)r * N + nt * 16 + n0] = out[nt][i];
        }
    }
}

// ---------------------------------------------------------------------------
// Fused row-wise MLP kernel (2 or 3 layers, SELU everywhere)
// MODE: 0 = direct x[rows][K0RAW]; 1 = gather emb[gidx[r]][K0RAW];
//       2 = phi_e concat(atoms[ba1],atoms[ba2],bonds,state)  (K0=128)
//       3 = phi_v concat(bta,atoms,state)                     (K0=96)
// ---------------------------------------------------------------------------
struct MlpArgs {
    const float* x;    const int* gidx;
    const float* srcA; const float* srcB; const float* srcS;
    const int* ba1;    const int* ba2;
    const float* W1;   const float* B1;
    const float* W2;   const float* B2;
    const float* W3;   const float* B3;
    float* out;        int rows;
};

template<int KRAW, int KP, int N>
__device__ __forceinline__ void stage_weights(unsigned short* wt, const float* W,
                                              const float* B, float* bs,
                                              int tid, int nthr) {
    for (int idx = tid; idx < KP * N; idx += nthr) {
        int k = idx / N, n = idx - k * N;
        float w = (k < KRAW) ? W[(size_t)k * N + n] : 0.f;
        wt[n * (KP + 8) + k] = f2bf(w);
    }
    for (int n = tid; n < N; n += nthr) bs[n] = B[n];
}

template<int MODE, int K0RAW, int K0P>
__device__ __forceinline__ void stage_tile(unsigned short* xt, int xs, int row0,
                                           const MlpArgs& A) {
    const int lane = threadIdx.x & 31;
    constexpr int CPR = K0P / 4;   // float4 chunks per row
    for (int idx = lane; idx < 16 * CPR; idx += 32) {
        int m = idx / CPR, c = idx - m * CPR;
        int r = row0 + m; if (r >= A.rows) r = A.rows - 1;
        int col = c * 4;
        float4v f = {0.f, 0.f, 0.f, 0.f};
        if constexpr (MODE == 0) {
            if (col < K0RAW) f = *(const float4v*)(A.x + (size_t)r * K0RAW + col);
        } else if constexpr (MODE == 1) {
            if (col < K0RAW) {
                int g = A.gidx[r];
                f = *(const float4v*)(A.x + (size_t)g * K0RAW + col);
            }
        } else if constexpr (MODE == 2) {
            if (col < 32)      f = *(const float4v*)(A.srcA + (size_t)A.ba1[r] * 32 + col);
            else if (col < 64) f = *(const float4v*)(A.srcA + (size_t)A.ba2[r] * 32 + (col - 32));
            else if (col < 96) f = *(const float4v*)(A.srcB + (size_t)r * 32 + (col - 64));
            else               f = *(const float4v*)(A.srcS + (col - 96));
        } else {
            if (col < 32)      f = *(const float4v*)(A.srcA + (size_t)r * 32 + col);
            else if (col < 64) f = *(const float4v*)(A.srcB + (size_t)r * 32 + (col - 32));
            else               f = *(const float4v*)(A.srcS + (col - 64));
        }
        uint2v pk;
        pk.x = (unsigned)f2bf(f.x) | ((unsigned)f2bf(f.y) << 16);
        pk.y = (unsigned)f2bf(f.z) | ((unsigned)f2bf(f.w) << 16);
        *(uint2v*)(xt + m * xs + col) = pk;
    }
}

template<int MODE, int NL, int K0RAW, int K0P, int H1, int H2, int NOUT>
__launch_bounds__(128)
__global__ void k_mlp(MlpArgs A) {
    constexpr int XS  = K0P + 8;
    constexpr int YS  = H1 + 8;
    constexpr int L2N = (NL == 3) ? H2 : NOUT;
    __shared__ alignas(16) unsigned short sW1[H1 * XS];
    __shared__ alignas(16) unsigned short sW2[L2N * YS];
    __shared__ alignas(16) unsigned short sW3[(NL == 3) ? NOUT * (H2 + 8) : 16];
    __shared__ float sB1[H1];
    __shared__ float sB2[L2N];
    __shared__ float sB3[(NL == 3) ? NOUT : 1];
    __shared__ alignas(16) unsigned short sX[4][16 * XS];
    __shared__ alignas(16) unsigned short sY[4][16 * YS];

    const int tid = threadIdx.x;
    stage_weights<K0RAW, K0P, H1>(sW1, A.W1, A.B1, sB1, tid, blockDim.x);
    stage_weights<H1, H1, L2N>(sW2, A.W2, A.B2, sB2, tid, blockDim.x);
    if constexpr (NL == 3)
        stage_weights<H2, H2, NOUT>(sW3, A.W3, A.B3, sB3, tid, blockDim.x);
    __syncthreads();

    const int wid    = tid >> 5;
    const int ntiles = (A.rows + 15) >> 4;
    for (int t = blockIdx.x * 4 + wid; t < ntiles; t += gridDim.x * 4) {
        const int row0 = t << 4;
        unsigned short* xt = sX[wid];
        unsigned short* yt = sY[wid];
        stage_tile<MODE, K0RAW, K0P>(xt, XS, row0, A);
        LDS_FENCE();
        float o1[H1 / 16][8];
        wave_gemm<K0P, H1>(xt, XS, sW1, XS, sB1, o1);
        wave_store_tile<H1>(yt, YS, o1);
        LDS_FENCE();
        if constexpr (NL == 2) {
            float o2[NOUT / 16][8];
            wave_gemm<H1, NOUT>(yt, YS, sW2, YS, sB2, o2);
            wave_store_global<NOUT>(A.out, row0, A.rows, o2);
        } else {
            float o2[H2 / 16][8];
            wave_gemm<H1, H2>(yt, YS, sW2, YS, sB2, o2);
            wave_store_tile<H2>(xt, XS, o2);      // reuse input tile
            LDS_FENCE();
            float o3[NOUT / 16][8];
            wave_gemm<H2, NOUT>(xt, XS, sW3, H2 + 8, sB3, o3);
            wave_store_global<NOUT>(A.out, row0, A.rows, o3);
        }
        LDS_FENCE();
    }
}

// ---------------------------------------------------------------------------
// Small scalar MLP (M <= 16, widths <= 256), up to 3 concat sources w/ scales
// ---------------------------------------------------------------------------
struct SmallArgs {
    const float* s0; int w0; float sc0; int b0;
    const float* s1; int w1; float sc1; int b1;
    const float* s2; int w2; float sc2; int b2;
    const float* W1; const float* B1; int h1;
    const float* W2; const float* B2; int h2;
    const float* W3; const float* B3; int nout;
    float* out; int M;
};

__launch_bounds__(256)
__global__ void k_small_mlp(SmallArgs A) {
    __shared__ float a0[16 * 256];
    __shared__ float a1[16 * 256];
    const int K0 = A.w0 + A.w1 + A.w2;
    for (int idx = threadIdx.x; idx < A.M * K0; idx += blockDim.x) {
        int m = idx / K0, c = idx - m * K0;
        float v;
        if (c < A.w0)               v = A.s0[(A.b0 ? 0 : m) * A.w0 + c] * A.sc0;
        else if (c < A.w0 + A.w1)   v = A.s1[(A.b1 ? 0 : m) * A.w1 + (c - A.w0)] * A.sc1;
        else                        v = A.s2[(A.b2 ? 0 : m) * A.w2 + (c - A.w0 - A.w1)] * A.sc2;
        a0[m * 256 + c] = v;
    }
    __syncthreads();
    const int nl = A.W3 ? 3 : 2;
    int dims0 = K0;
    int dims1 = A.h1;
    int dims2 = (nl == 3) ? A.h2 : A.nout;
    int dims3 = A.nout;
    for (int L = 0; L < nl; ++L) {
        // ping-pong LDS buffers selected by ternary (no LDS-pointer arrays:
        // those produce unsupported addrspacecast static initializers)
        float* x = (L & 1) ? a1 : a0;
        float* y = (L & 1) ? a0 : a1;
        const float* W = (L == 0) ? A.W1 : (L == 1) ? A.W2 : A.W3;
        const float* B = (L == 0) ? A.B1 : (L == 1) ? A.B2 : A.B3;
        const int K = (L == 0) ? dims0 : (L == 1) ? dims1 : dims2;
        const int N = (L == 0) ? dims1 : (L == 1) ? dims2 : dims3;
        for (int idx = threadIdx.x; idx < A.M * N; idx += blockDim.x) {
            int m = idx / N, n = idx - m * N;
            float s = B[n];
            const float* xr = x + m * 256;
            const float* Wc = W + n;
            for (int k = 0; k < K; ++k) s += xr[k] * Wc[(size_t)k * N];
            y[m * 256 + n] = selu_f(s);
        }
        __syncthreads();
    }
    const float* res = (nl & 1) ? a1 : a0;
    const int N = (nl == 3) ? dims3 : dims2;
    for (int idx = threadIdx.x; idx < A.M * N; idx += blockDim.x)
        A.out[idx] = res[(idx / N) * 256 + (idx % N)];
}

// ---------------------------------------------------------------------------
// Elementwise / scatter / reduction kernels
// ---------------------------------------------------------------------------
__global__ void k_add(float* dst, const float* src, size_t n) {
    size_t stride = (size_t)gridDim.x * blockDim.x;
    for (size_t i = (size_t)blockIdx.x * blockDim.x + threadIdx.x; i < n; i += stride)
        dst[i] += src[i];
}

__global__ void k_cnt(const int* ba1, float* cnt, int nb) {
    int i = blockIdx.x * blockDim.x + threadIdx.x;
    if (i < nb) atomicAdd(cnt + ba1[i], 1.f);
}

__global__ void k_scatter(const float* rb, const int* ba1, float* bta, size_t n32) {
    size_t stride = (size_t)gridDim.x * blockDim.x;
    for (size_t i = (size_t)blockIdx.x * blockDim.x + threadIdx.x; i < n32; i += stride)
        atomicAdd(&bta[(size_t)ba1[i >> 5] * 32 + (i & 31)], rb[i]);
}

__global__ void k_div(float* bta, const float* cnt, size_t n32) {
    size_t i = (size_t)blockIdx.x * blockDim.x + threadIdx.x;
    if (i < n32) bta[i] /= cnt[i >> 5];
}

__global__ void k_colsum(const float* x, int n, float* out32) {
    const int col = threadIdx.x & 31;
    int g = (blockIdx.x * blockDim.x + threadIdx.x) >> 5;
    const int stride = (gridDim.x * blockDim.x) >> 5;
    float s = 0.f;
    for (int i = g; i < n; i += stride) s += x[(size_t)i * 32 + col];
    atomicAdd(out32 + col, s);
}

// ---------------------------------------------------------------------------
// Set2Set: LSTM step + attention pooling (d = 32, 16 graphs)
// ---------------------------------------------------------------------------
__launch_bounds__(256)
__global__ void k_lstm(const float* Wih, const float* Whh, const float* bih,
                       const float* bhh, const float* qstar, float* h, float* c,
                       unsigned* emax, float* asum, float* r) {
    __shared__ float g[16 * 128];
    const int tid = threadIdx.x;
    if (tid < 16) { emax[tid] = 0u; asum[tid] = 0.f; }      // reset attn scratch
    for (int i = tid; i < 16 * 32; i += blockDim.x) r[i] = 0.f;
    for (int idx = tid; idx < 16 * 128; idx += blockDim.x) {
        int m = idx >> 7, n = idx & 127;
        float s = bih[n] + bhh[n];
        for (int k = 0; k < 64; ++k) s += qstar[m * 64 + k] * Wih[k * 128 + n];
        for (int k = 0; k < 32; ++k) s += h[m * 32 + k] * Whh[k * 128 + n];
        g[idx] = s;
    }
    __syncthreads();
    for (int idx = tid; idx < 16 * 32; idx += blockDim.x) {
        int m = idx >> 5, j = idx & 31;
        float ig = sigm_f(g[m * 128 + j]);
        float fg = sigm_f(g[m * 128 + 32 + j]);
        float gg = tanhf(g[m * 128 + 64 + j]);
        float og = sigm_f(g[m * 128 + 96 + j]);
        float cc = fg * c[idx] + ig * gg;
        c[idx] = cc;
        h[idx] = og * tanhf(cc);
    }
}

__global__ void k_attn_e(const float* x, const float* h, const int* batch, int n,
                         float* e, unsigned* emax) {
    int i = blockIdx.x * blockDim.x + threadIdx.x;
    if (i >= n) return;
    int b = batch[i];
    const float* xr = x + (size_t)i * 32;
    const float* q = h + b * 32;
    float s = 0.f;
#pragma unroll
    for (int c = 0; c < 32; ++c) s += xr[c] * q[c];
    e[i] = s;
    unsigned u = __float_as_uint(s);
    u = (u & 0x80000000u) ? ~u : (u | 0x80000000u);   // order-preserving encode
    atomicMax(emax + b, u);
}

__global__ void k_attn_exp(const float* e, const int* batch, int n,
                           const unsigned* emax, float* a, float* asum) {
    int i = blockIdx.x * blockDim.x + threadIdx.x;
    if (i >= n) return;
    int b = batch[i];
    unsigned u = emax[b];
    u = (u & 0x80000000u) ? (u & 0x7FFFFFFFu) : ~u;
    float av = __expf(e[i] - __uint_as_float(u));
    a[i] = av;
    atomicAdd(asum + b, av);
}

__launch_bounds__(256)
__global__ void k_attn_r(const float* x, const float* a, const float* asum,
                         const int* batch, int n, float* r) {
    __shared__ float rl[512];
    for (int i = threadIdx.x; i < 512; i += blockDim.x) rl[i] = 0.f;
    __syncthreads();
    const int pairsPerBlock = blockDim.x * 128;
    const size_t base = (size_t)blockIdx.x * pairsPerBlock;
    const size_t total = (size_t)n * 32;
    for (int t = threadIdx.x; t < pairsPerBlock; t += blockDim.x) {
        size_t idx = base + t;
        if (idx >= total) break;
        int i = (int)(idx >> 5), cc = (int)(idx & 31);
        int b = batch[i];
        float w = a[i] / asum[b];
        atomicAdd(&rl[b * 32 + cc], w * x[idx]);
    }
    __syncthreads();
    for (int i = threadIdx.x; i < 512; i += blockDim.x) atomicAdd(&r[i], rl[i]);
}

__global__ void k_qstar(const float* h, const float* r, float* qstar) {
    int idx = blockIdx.x * blockDim.x + threadIdx.x;
    if (idx < 16 * 64) {
        int m = idx >> 6, c = idx & 63;
        qstar[idx] = (c < 32) ? h[m * 32 + c] : r[m * 32 + c - 32];
    }
}

// ---------------------------------------------------------------------------
// Host orchestration
// ---------------------------------------------------------------------------
extern "C" void kernel_launch(void* const* d_in, const int* in_sizes, int n_in,
                              void* d_out, int out_size, void* d_ws, size_t ws_size,
                              hipStream_t stream) {
    (void)n_in; (void)out_size; (void)ws_size;
    const int*   atoms = (const int*)d_in[0];
    const float* state = (const float*)d_in[1];
    const float* bonds = (const float*)d_in[2];
    const int*   ba1   = (const int*)d_in[3];
    const int*   ba2   = (const int*)d_in[4];
    const int*   bm_a  = (const int*)d_in[5];
    const int*   bm_b  = (const int*)d_in[6];
    const int NA = in_sizes[0];
    const int NB = in_sizes[3];

    // params leaves in jax tree order (dict keys sorted recursively):
    //   0:atom_pre(4)  4:block0.layer phi_e/phi_u/phi_v (18)
    //   22+(b-1)*30: blocks 1..3 = atoms_ff(4), bonds_ff(4), layer(18), state_ff(4)
    //   112:bond_pre(4) 116:emb 117:output(6)
    //   123:s2s_e Whh,Wih,bhh,bih  127:s2s_v  131:state_pre(4)
    auto P = [&](int i) -> const float* { return (const float*)d_in[7 + i]; };

    uint8_t* wsb = (uint8_t*)d_ws;
    size_t off = 0;
    auto alloc = [&](size_t bytes) -> void* {
        void* p = wsb + off;
        off += (bytes + 255) & ~(size_t)255;
        return p;
    };
    float* atomsF = (float*)alloc((size_t)NA * 32 * 4);
    float* bondsF = (float*)alloc((size_t)NB * 32 * 4);
    float* ffa    = (float*)alloc((size_t)NA * 32 * 4);
    float* ffb    = (float*)alloc((size_t)NB * 32 * 4);
    float* rb     = (float*)alloc((size_t)NB * 32 * 4);
    float* ra     = (float*)alloc((size_t)NA * 32 * 4);
    float* bta    = (float*)alloc((size_t)NA * 32 * 4);
    float* cnt    = (float*)alloc((size_t)NA * 4);
    float* stateF = (float*)alloc(256);
    float* ffs    = (float*)alloc(256);
    float* rs     = (float*)alloc(256);
    float* meanb  = (float*)alloc(256);
    float* meana  = (float*)alloc(256);
    float* h_     = (float*)alloc(16 * 32 * 4);
    float* c_     = (float*)alloc(16 * 32 * 4);
    float* rS     = (float*)alloc(16 * 32 * 4);
    float* asum_  = (float*)alloc(256);
    unsigned* emaxu = (unsigned*)alloc(256);
    float* qsE    = (float*)alloc(16 * 64 * 4);
    float* qsV    = (float*)alloc(16 * 64 * 4);
    float* e_  = rb;            // alias: rb is dead by set2set time
    float* aat = rb + NB;

    const int nblkA = ((NA + 15) / 16 + 3) / 4;
    const int nblkB = ((NB + 15) / 16 + 3) / 4;

    // --- preludes -----------------------------------------------------------
    { MlpArgs a{}; a.x = P(116); a.gidx = atoms;
      a.W1 = P(0); a.B1 = P(1); a.W2 = P(2); a.B2 = P(3);
      a.out = atomsF; a.rows = NA;
      k_mlp<1, 2, 16, 32, 64, 16, 32><<<nblkA, 128, 0, stream>>>(a); }
    { MlpArgs a{}; a.x = bonds;
      a.W1 = P(112); a.B1 = P(113); a.W2 = P(114); a.B2 = P(115);
      a.out = bondsF; a.rows = NB;
      k_mlp<0, 2, 100, 128, 64, 16, 32><<<nblkB, 128, 0, stream>>>(a); }
    { SmallArgs s{}; s.s0 = state; s.w0 = 2; s.sc0 = 1.f;
      s.W1 = P(131); s.B1 = P(132); s.h1 = 64;
      s.W2 = P(133); s.B2 = P(134); s.nout = 32; s.W3 = nullptr;
      s.out = stateF; s.M = 1;
      k_small_mlp<<<1, 256, 0, stream>>>(s); }
    (void)hipMemsetAsync(cnt, 0, (size_t)NA * 4, stream);
    k_cnt<<<(NB + 255) / 256, 256, 0, stream>>>(ba1, cnt, NB);

    // --- 4 MEGNet blocks ----------------------------------------------------
    for (int b = 0; b < 4; ++b) {
        const float *inB, *inA, *inS; int Lb;
        if (b == 0) { inB = bondsF; inA = atomsF; inS = stateF; Lb = 4; }
        else {
            int bb = 22 + (b - 1) * 30;
            { MlpArgs a{}; a.x = atomsF;
              a.W1 = P(bb + 0); a.B1 = P(bb + 1); a.W2 = P(bb + 2); a.B2 = P(bb + 3);
              a.out = ffa; a.rows = NA;
              k_mlp<0, 2, 32, 32, 64, 16, 32><<<nblkA, 128, 0, stream>>>(a); }
            { MlpArgs a{}; a.x = bondsF;
              a.W1 = P(bb + 4); a.B1 = P(bb + 5); a.W2 = P(bb + 6); a.B2 = P(bb + 7);
              a.out = ffb; a.rows = NB;
              k_mlp<0, 2, 32, 32, 64, 16, 32><<<nblkB, 128, 0, stream>>>(a); }
            { SmallArgs s{}; s.s0 = stateF; s.w0 = 32; s.sc0 = 1.f;
              s.W1 = P(bb + 26); s.B1 = P(bb + 27); s.h1 = 64;
              s.W2 = P(bb + 28); s.B2 = P(bb + 29); s.nout = 32; s.W3 = nullptr;
              s.out = ffs; s.M = 1;
              k_small_mlp<<<1, 256, 0, stream>>>(s); }
            inB = ffb; inA = ffa; inS = ffs; Lb = bb + 8;
        }
        // phi_e -> rb
        { MlpArgs a{}; a.srcA = inA; a.srcB = inB; a.srcS = inS; a.ba1 = ba1; a.ba2 = ba2;
          a.W1 = P(Lb + 0); a.B1 = P(Lb + 1); a.W2 = P(Lb + 2); a.B2 = P(Lb + 3);
          a.W3 = P(Lb + 4); a.B3 = P(Lb + 5);
          a.out = rb; a.rows = NB;
          k_mlp<2, 3, 128, 128, 64, 64, 32><<<nblkB, 128, 0, stream>>>(a); }
        // scatter-mean(rb by ba1) -> bta
        (void)hipMemsetAsync(bta, 0, (size_t)NA * 32 * 4, stream);
        k_scatter<<<4096, 256, 0, stream>>>(rb, ba1, bta, (size_t)NB * 32);
        k_div<<<(int)(((size_t)NA * 32 + 255) / 256), 256, 0, stream>>>(bta, cnt, (size_t)NA * 32);
        // phi_v -> ra
        { MlpArgs a{}; a.srcA = bta; a.srcB = inA; a.srcS = inS;
          a.W1 = P(Lb + 12); a.B1 = P(Lb + 13); a.W2 = P(Lb + 14); a.B2 = P(Lb + 15);
          a.W3 = P(Lb + 16); a.B3 = P(Lb + 17);
          a.out = ra; a.rows = NA;
          k_mlp<3, 3, 96, 96, 64, 64, 32><<<nblkA, 128, 0, stream>>>(a); }
        // means + phi_u -> rs
        (void)hipMemsetAsync(meanb, 0, 128, stream);
        (void)hipMemsetAsync(meana, 0, 128, stream);
        k_colsum<<<256, 256, 0, stream>>>(rb, NB, meanb);
        k_colsum<<<256, 256, 0, stream>>>(ra, NA, meana);
        { SmallArgs s{}; s.s0 = meanb; s.w0 = 32; s.sc0 = 1.f / (float)NB;
          s.s1 = meana; s.w1 = 32; s.sc1 = 1.f / (float)NA;
          s.s2 = inS;   s.w2 = 32; s.sc2 = 1.f;
          s.W1 = P(Lb + 6);  s.B1 = P(Lb + 7);  s.h1 = 64;
          s.W2 = P(Lb + 8);  s.B2 = P(Lb + 9);  s.h2 = 64;
          s.W3 = P(Lb + 10); s.B3 = P(Lb + 11); s.nout = 32;
          s.out = rs; s.M = 1;
          k_small_mlp<<<1, 256, 0, stream>>>(s); }
        // residual adds
        k_add<<<2048, 256, 0, stream>>>(bondsF, rb, (size_t)NB * 32);
        k_add<<<512, 256, 0, stream>>>(atomsF, ra, (size_t)NA * 32);
        k_add<<<1, 64, 0, stream>>>(stateF, rs, 32);
    }

    // --- set2set pooling ----------------------------------------------------
    auto set2set = [&](const float* x, const int* batch, int n, int lb, float* qstar) {
        (void)hipMemsetAsync(h_, 0, 16 * 32 * 4, stream);
        (void)hipMemsetAsync(c_, 0, 16 * 32 * 4, stream);
        (void)hipMemsetAsync(qstar, 0, 16 * 64 * 4, stream);
        const float* Whh = P(lb + 0); const float* Wih = P(lb + 1);
        const float* bhh = P(lb + 2); const float* bih = P(lb + 3);
        const int nb1 = (n + 255) / 256;
        const int pairsB = 256 * 128;
        const int nb2 = (int)(((size_t)n * 32 + pairsB - 1) / pairsB);
        for (int s = 0; s < 3; ++s) {
            k_lstm<<<1, 256, 0, stream>>>(Wih, Whh, bih, bhh, qstar, h_, c_, emaxu, asum_, rS);
            k_attn_e<<<nb1, 256, 0, stream>>>(x, h_, batch, n, e_, emaxu);
            k_attn_exp<<<nb1, 256, 0, stream>>>(e_, batch, n, emaxu, aat, asum_);
            k_attn_r<<<nb2, 256, 0, stream>>>(x, aat, asum_, batch, n, rS);
            k_qstar<<<4, 256, 0, stream>>>(h_, rS, qstar);
        }
    };
    set2set(bondsF, bm_b, NB, 123, qsE);
    set2set(atomsF, bm_a, NA, 127, qsV);

    // --- output head --------------------------------------------------------
    { SmallArgs s{}; s.s0 = qsE; s.w0 = 64; s.sc0 = 1.f;
      s.s1 = qsV; s.w1 = 64; s.sc1 = 1.f;
      s.s2 = stateF; s.w2 = 32; s.sc2 = 1.f; s.b2 = 1;
      s.W1 = P(117); s.B1 = P(118); s.h1 = 128;
      s.W2 = P(119); s.B2 = P(120); s.h2 = 64;
      s.W3 = P(121); s.B3 = P(122); s.nout = 200;
      s.out = (float*)d_out; s.M = 16;
      k_small_mlp<<<1, 256, 0, stream>>>(s); }
}
